// Self_Attention_Head_738734375637
// MI455X (gfx1250) — compile-verified, matrix operand order checked by reading, not run on hardware
//
#include <hip/hip_runtime.h>
#include <hip/hip_bf16.h>

#define B_ 4
#define S_ 2048
#define D_ 1024
#define H_ 16
#define HD_ 64
#define M_ (B_ * S_)   // 8192 rows of the token matrix

typedef __attribute__((ext_vector_type(16))) __bf16 bf16x16;
typedef __attribute__((ext_vector_type(8)))  float  f32x8;

union Frag {          // 16 bf16 = 8 VGPRs; u[0] -> K = g*8..g*8+7, u[1] -> K = 16+g*8..
    uint4   u[2];
    bf16x16 v;
};

#define WMMA_BF16(a, b, c) \
    __builtin_amdgcn_wmma_f32_16x16x32_bf16(false, (a), false, (b), (short)0, (c), false, false)

// ---------------------------------------------------------------------------
// fp32 -> bf16 conversion (4 elems / thread)
// ---------------------------------------------------------------------------
__global__ void f32_to_bf16_kernel(const float* __restrict__ in, __bf16* __restrict__ out, int n) {
    int i = (blockIdx.x * blockDim.x + threadIdx.x) * 4;
    if (i + 3 < n) {
        float4 f = *reinterpret_cast<const float4*>(in + i);
        out[i + 0] = (__bf16)f.x;
        out[i + 1] = (__bf16)f.y;
        out[i + 2] = (__bf16)f.z;
        out[i + 3] = (__bf16)f.w;
    } else {
        for (int j = i; j < n; ++j) out[j] = (__bf16)in[j];
    }
}

// ---------------------------------------------------------------------------
// NT GEMM: C[M x N] = A[M x K](bf16) * W[N x K](bf16)^T  (+bias)*scale (+res)
// block = 256 threads (8 waves); block tile 128x128; wave tile 32x64
// (2 M-tiles x 4 N-tiles = 8 f32x8 accumulators, 6 fragment loads / 8 WMMAs).
// W tile double-buffered through LDS; one barrier per 32-K step.
// MODE 0: bf16 output.  MODE 1: f32 output + residual add.
// ---------------------------------------------------------------------------
template <int MODE>
__global__ __launch_bounds__(256)
void gemm_nt_kernel(const __bf16* __restrict__ A, const __bf16* __restrict__ W,
                    const float* __restrict__ bias, void* __restrict__ Out,
                    const float* __restrict__ residual,
                    int M, int N, int K, float scale) {
    __shared__ __attribute__((aligned(16))) __bf16 wlds[2][128 * 40];  // padded stride 40

    const int lane = threadIdx.x & 31;
    const int wave = threadIdx.x >> 5;
    const int g    = lane >> 4;      // half-wave group (A/B layout K-offset selector)
    const int lm   = lane & 15;      // row (A) / column (B) within tile
    const int mw   = wave & 3;       // 32-row strip within block tile
    const int nw   = wave >> 2;      // 64-col strip within block tile
    const int n0   = blockIdx.x * 128;
    const int m0   = blockIdx.y * 128;

    f32x8 acc[2][4];
    for (int mt = 0; mt < 2; ++mt)
        for (int j = 0; j < 4; ++j)
            for (int e = 0; e < 8; ++e) acc[mt][j][e] = 0.0f;

    // cooperative W staging: 128 rows x 2 threads, 32 B each
    const int srow  = threadIdx.x >> 1;
    const int shalf = threadIdx.x & 1;
    const __bf16* wstage = W + (size_t)(n0 + srow) * K + shalf * 16;

    // preload iteration 0 into buffer 0
    {
        uint4 w0 = *reinterpret_cast<const uint4*>(wstage);
        uint4 w1 = *reinterpret_cast<const uint4*>(wstage + 8);
        *reinterpret_cast<uint4*>(&wlds[0][srow * 40 + shalf * 16])     = w0;
        *reinterpret_cast<uint4*>(&wlds[0][srow * 40 + shalf * 16 + 8]) = w1;
    }

    const int niter = K / 32;
    for (int i = 0; i < niter; ++i) {
        __syncthreads();                       // buffer (i&1) is ready
        const int kk = i * 32;

        uint4 nx0, nx1;
        const bool havenext = (i + 1 < niter);
        if (havenext) {                        // issue next tile's global loads early
            nx0 = *reinterpret_cast<const uint4*>(wstage + kk + 32);
            nx1 = *reinterpret_cast<const uint4*>(wstage + kk + 32 + 8);
            if (i + 2 < niter)
                __builtin_prefetch(wstage + kk + 64, 0, 1);  // global_prefetch_b8
        }

        Frag fa[2];
#pragma unroll
        for (int mt = 0; mt < 2; ++mt) {
            const __bf16* asrc = A + (size_t)(m0 + mw * 32 + mt * 16 + lm) * K + kk;
            fa[mt].u[0] = *reinterpret_cast<const uint4*>(asrc + g * 8);
            fa[mt].u[1] = *reinterpret_cast<const uint4*>(asrc + 16 + g * 8);
        }

        const __bf16* wl = wlds[i & 1];
#pragma unroll
        for (int j = 0; j < 4; ++j) {
            Frag fb;
            const __bf16* bsrc = &wl[(nw * 64 + j * 16 + lm) * 40];
            fb.u[0] = *reinterpret_cast<const uint4*>(bsrc + g * 8);
            fb.u[1] = *reinterpret_cast<const uint4*>(bsrc + 16 + g * 8);
            acc[0][j] = WMMA_BF16(fa[0].v, fb.v, acc[0][j]);
            acc[1][j] = WMMA_BF16(fa[1].v, fb.v, acc[1][j]);
        }

        if (havenext) {                        // fill the idle buffer
            __bf16* dl = wlds[(i + 1) & 1];
            *reinterpret_cast<uint4*>(&dl[srow * 40 + shalf * 16])     = nx0;
            *reinterpret_cast<uint4*>(&dl[srow * 40 + shalf * 16 + 8]) = nx1;
        }
    }

    // epilogue: C-layout row m = r + 8*g, col n = lm
#pragma unroll
    for (int mt = 0; mt < 2; ++mt) {
        const int mbase = m0 + mw * 32 + mt * 16 + 8 * g;
#pragma unroll
        for (int j = 0; j < 4; ++j) {
            const int n  = n0 + nw * 64 + j * 16 + lm;
            const float bv = bias ? bias[n] : 0.0f;
#pragma unroll
            for (int r = 0; r < 8; ++r) {
                const int m = mbase + r;
                float v = (acc[mt][j][r] + bv) * scale;
                if (MODE == 0) {
                    ((__bf16*)Out)[(size_t)m * N + n] = (__bf16)v;
                } else {
                    v += residual[(size_t)m * N + n];
                    ((float*)Out)[(size_t)m * N + n] = v;
                }
            }
        }
    }
}

// ---------------------------------------------------------------------------
// Flash attention: scores = Q_scaled * K^T + tril(1); softmax; ctx = A * V.
// grid = (S/128, B*H); block = 256 (8 waves); wave owns 16 query rows.
// K/V tiles double-buffered in LDS; one barrier per 32-key step.
// Q/K/V/ctx are (B,S,D) bf16 with head h in columns [h*64, h*64+64).
// ---------------------------------------------------------------------------
__global__ __launch_bounds__(256)
void flash_attn_kernel(const __bf16* __restrict__ Q, const __bf16* __restrict__ Kp,
                       const __bf16* __restrict__ Vp, __bf16* __restrict__ O) {
    __shared__ __attribute__((aligned(16))) __bf16 klds[2][32 * 72];     // K tile row-major
    __shared__ __attribute__((aligned(16))) __bf16 vlds[2][64 * 40];     // V tile TRANSPOSED [d][key]
    __shared__ __attribute__((aligned(16))) __bf16 plds[8 * 16 * 40];    // per-wave P (16x32)

    const int lane = threadIdx.x & 31;
    const int wave = threadIdx.x >> 5;
    const int g    = lane >> 4;
    const int lm   = lane & 15;

    const int b = blockIdx.y / H_;
    const int h = blockIdx.y % H_;
    const int qblk = blockIdx.x * 128;
    const size_t headoff = (size_t)b * S_ * D_ + (size_t)h * HD_;

    // Q fragments (two K=32 steps over HD=64), loaded once
    const int qrow = qblk + wave * 16 + lm;
    Frag qf[2];
    const __bf16* qsrc = Q + headoff + (size_t)qrow * D_;
#pragma unroll
    for (int dh = 0; dh < 2; ++dh) {
        qf[dh].u[0] = *reinterpret_cast<const uint4*>(qsrc + dh * 32 + g * 8);
        qf[dh].u[1] = *reinterpret_cast<const uint4*>(qsrc + dh * 32 + 16 + g * 8);
    }

    float mrun[8], lrun[8];
    f32x8 acc[4];
    for (int r = 0; r < 8; ++r) { mrun[r] = -3.0e38f; lrun[r] = 0.0f; }
    for (int dt = 0; dt < 4; ++dt)
        for (int e = 0; e < 8; ++e) acc[dt][e] = 0.0f;

    // staging: 32 key rows x 8 d-chunks, 16 B each
    const int trow = threadIdx.x >> 3;
    const int tch  = threadIdx.x & 7;
    const __bf16* kstage = Kp + headoff + (size_t)trow * D_ + tch * 8;
    const __bf16* vstage = Vp + headoff + (size_t)trow * D_ + tch * 8;

    // preload tile 0 into buffer 0
    {
        uint4 kv = *reinterpret_cast<const uint4*>(kstage);
        union { uint4 u; __bf16 e[8]; } vv;
        vv.u = *reinterpret_cast<const uint4*>(vstage);
        *reinterpret_cast<uint4*>(&klds[0][trow * 72 + tch * 8]) = kv;
#pragma unroll
        for (int i = 0; i < 8; ++i) vlds[0][(tch * 8 + i) * 40 + trow] = vv.e[i];
    }

    const int niter = S_ / 32;
    for (int it = 0; it < niter; ++it) {
        __syncthreads();                         // buffer (it&1) is ready
        const int kt = it * 32;

        uint4 kn;
        union { uint4 u; __bf16 e[8]; } vn;
        const bool havenext = (it + 1 < niter);
        if (havenext) {
            kn   = *reinterpret_cast<const uint4*>(kstage + (size_t)(kt + 32) * D_);
            vn.u = *reinterpret_cast<const uint4*>(vstage + (size_t)(kt + 32) * D_);
        }

        const __bf16* kl = klds[it & 1];
        const __bf16* vl = vlds[it & 1];

        // scores: two 16-key subtiles, accumulated over two d-halves
        f32x8 sc[2];
        for (int t = 0; t < 2; ++t)
            for (int e = 0; e < 8; ++e) sc[t][e] = 0.0f;
#pragma unroll
        for (int dh = 0; dh < 2; ++dh) {
#pragma unroll
            for (int t = 0; t < 2; ++t) {
                Frag kf;
                const __bf16* ks = &kl[(t * 16 + lm) * 72 + dh * 32];
                kf.u[0] = *reinterpret_cast<const uint4*>(ks + g * 8);
                kf.u[1] = *reinterpret_cast<const uint4*>(ks + 16 + g * 8);
                sc[t] = WMMA_BF16(qf[dh].v, kf.v, sc[t]);
            }
        }

        // additive tril(1) mask + online softmax (row = r + 8*g, 16 lanes per row)
        __bf16* pbase = &plds[wave * 16 * 40];
#pragma unroll
        for (int r = 0; r < 8; ++r) {
            const int qg = qblk + wave * 16 + r + 8 * g;
            float s0 = sc[0][r] + ((kt + lm)      <= qg ? 1.0f : 0.0f);
            float s1 = sc[1][r] + ((kt + 16 + lm) <= qg ? 1.0f : 0.0f);
            float mx = fmaxf(s0, s1);
            for (int off = 1; off < 16; off <<= 1)
                mx = fmaxf(mx, __shfl_xor(mx, off, 32));
            const float mnew  = fmaxf(mrun[r], mx);
            const float alpha = __expf(mrun[r] - mnew);
            const float p0 = __expf(s0 - mnew);
            const float p1 = __expf(s1 - mnew);
            float rs = p0 + p1;
            for (int off = 1; off < 16; off <<= 1)
                rs += __shfl_xor(rs, off, 32);
            lrun[r] = lrun[r] * alpha + rs;
            mrun[r] = mnew;
#pragma unroll
            for (int dt = 0; dt < 4; ++dt) acc[dt][r] *= alpha;
            pbase[(r + 8 * g) * 40 + lm]      = (__bf16)p0;
            pbase[(r + 8 * g) * 40 + 16 + lm] = (__bf16)p1;
        }

        // P back out of LDS in A-operand layout (per-wave buffer, DS ops in-order)
        Frag pf;
        const __bf16* ps = &plds[wave * 16 * 40 + lm * 40];
        pf.u[0] = *reinterpret_cast<const uint4*>(ps + g * 8);
        pf.u[1] = *reinterpret_cast<const uint4*>(ps + 16 + g * 8);

#pragma unroll
        for (int dt = 0; dt < 4; ++dt) {
            Frag vf;
            const __bf16* vs = &vl[(dt * 16 + lm) * 40];
            vf.u[0] = *reinterpret_cast<const uint4*>(vs + g * 8);
            vf.u[1] = *reinterpret_cast<const uint4*>(vs + 16 + g * 8);
            acc[dt] = WMMA_BF16(pf.v, vf.v, acc[dt]);
        }

        if (havenext) {                          // fill the idle buffer
            __bf16* kd = klds[(it + 1) & 1];
            __bf16* vd = vlds[(it + 1) & 1];
            *reinterpret_cast<uint4*>(&kd[trow * 72 + tch * 8]) = kn;
#pragma unroll
            for (int i = 0; i < 8; ++i) vd[(tch * 8 + i) * 40 + trow] = vn.e[i];
        }
    }

    // normalize and store ctx (transposed back to (b,s,h*64+d) automatically)
    __bf16* dst = O + headoff;
#pragma unroll
    for (int dt = 0; dt < 4; ++dt)
#pragma unroll
        for (int r = 0; r < 8; ++r) {
            const int m = qblk + wave * 16 + r + 8 * g;
            dst[(size_t)m * D_ + dt * 16 + lm] = (__bf16)(acc[dt][r] / lrun[r]);
        }
}

// ---------------------------------------------------------------------------
extern "C" void kernel_launch(void* const* d_in, const int* in_sizes, int n_in,
                              void* d_out, int out_size, void* d_ws, size_t ws_size,
                              hipStream_t stream) {
    const float* x      = (const float*)d_in[0];
    const float* Wq     = (const float*)d_in[1];
    const float* Wk     = (const float*)d_in[2];
    const float* Wv     = (const float*)d_in[3];
    const float* in_w   = (const float*)d_in[4];
    const float* in_b   = (const float*)d_in[5];
    const float* out_w  = (const float*)d_in[6];
    const float* out_b  = (const float*)d_in[7];
    const float* proj_w = (const float*)d_in[8];
    const float* proj_b = (const float*)d_in[9];
    float* out = (float*)d_out;

    const size_t MB = 1u << 20;
    char* ws = (char*)d_ws;
    __bf16* xb     = (__bf16*)(ws + 0);         // 16 MB  (B*S*D)
    __bf16* wqb    = (__bf16*)(ws + 16 * MB);   //  2 MB
    __bf16* wkb    = (__bf16*)(ws + 18 * MB);
    __bf16* wvb    = (__bf16*)(ws + 20 * MB);
    __bf16* inwb   = (__bf16*)(ws + 22 * MB);   //  6 MB
    __bf16* outwb  = (__bf16*)(ws + 28 * MB);
    __bf16* projwb = (__bf16*)(ws + 30 * MB);
    __bf16* qb     = (__bf16*)(ws + 32 * MB);   // 16 MB each
    __bf16* kb     = (__bf16*)(ws + 48 * MB);
    __bf16* vb     = (__bf16*)(ws + 64 * MB);
    __bf16* Qb     = (__bf16*)(ws + 80 * MB);
    __bf16* Kb     = (__bf16*)(ws + 96 * MB);
    __bf16* Vb     = (__bf16*)(ws + 112 * MB);
    __bf16* ctx    = qb;                        // q/k/v dead after Q/K/V built
    __bf16* attn   = kb;

    const int NX = M_ * D_;      // 8,388,608
    const int ND = D_ * D_;      // 1,048,576
    auto cvt = [&](const float* src, __bf16* dst, int n) {
        f32_to_bf16_kernel<<<(n / 4 + 255) / 256, 256, 0, stream>>>(src, dst, n);
    };
    cvt(x, xb, NX);
    cvt(Wq, wqb, ND);  cvt(Wk, wkb, ND);  cvt(Wv, wvb, ND);
    cvt(in_w, inwb, 3 * ND);
    cvt(out_w, outwb, ND);  cvt(proj_w, projwb, ND);

    const dim3 blk(256);
    const dim3 gg(D_ / 128, M_ / 128);  // (8, 64)
    const float qscale = 1.0f / 8.0f;   // 1/sqrt(HD)

    // q/k/v = x * W^T
    gemm_nt_kernel<0><<<gg, blk, 0, stream>>>(xb, wqb, nullptr, qb, nullptr, M_, D_, D_, 1.0f);
    gemm_nt_kernel<0><<<gg, blk, 0, stream>>>(xb, wkb, nullptr, kb, nullptr, M_, D_, D_, 1.0f);
    gemm_nt_kernel<0><<<gg, blk, 0, stream>>>(xb, wvb, nullptr, vb, nullptr, M_, D_, D_, 1.0f);
    // Q/K/V = (q/k/v) * in_w_slice^T + bias   (softmax scale folded into Q)
    gemm_nt_kernel<0><<<gg, blk, 0, stream>>>(qb, inwb,          in_b,          Qb, nullptr, M_, D_, D_, qscale);
    gemm_nt_kernel<0><<<gg, blk, 0, stream>>>(kb, inwb + ND,     in_b + D_,     Kb, nullptr, M_, D_, D_, 1.0f);
    gemm_nt_kernel<0><<<gg, blk, 0, stream>>>(vb, inwb + 2 * ND, in_b + 2 * D_, Vb, nullptr, M_, D_, D_, 1.0f);
    // attention
    flash_attn_kernel<<<dim3(S_ / 128, B_ * H_), blk, 0, stream>>>(Qb, Kb, Vb, ctx);
    // attn = ctx * out_w^T + out_b
    gemm_nt_kernel<0><<<gg, blk, 0, stream>>>(ctx, outwb, out_b, attn, nullptr, M_, D_, D_, 1.0f);
    // out = attn * proj_w^T + proj_b + x   (fp32)
    gemm_nt_kernel<1><<<gg, blk, 0, stream>>>(attn, projwb, proj_b, out, x, M_, D_, D_, 1.0f);
}
